// ExpertsLayer_7387343749155
// MI455X (gfx1250) — matrix operand order repaired and checked by reading, not compiled, not run on hardware
//
#include <hip/hip_runtime.h>
#include <hip/hip_bf16.h>

// ---------------------------------------------------------------------------
// MoE expert layer: per-expert (Linear -> ReLU -> LayerNorm), equal capacity.
// T=8192 tokens, D=1024, E=8, cap=1024.
//
// MI455X (gfx1250, wave32) strategy:
//   1) swizzle_w_kernel: W f32 -> f16 in WMMA-B-fragment-native layout (d_ws,
//      16 MB). Amortizes f32->f16 conversion; makes every B-fragment load a
//      coalesced 32 B/lane global_load_b128 pair.
//   2) moe_fused_kernel: 32 tokens/block (M=32 doubles intensity vs L2 and
//      halves weight re-reads to 512 MB), 8 waves; each wave computes a
//      32x128 f32 tile via v_wmma_f32_16x16x32_f16 (2 WMMAs per B-fragment),
//      K staged to LDS as f16 in two 512-wide phases, then fused
//      bias+ReLU+LayerNorm epilogue (shfl reduce + LDS cross-wave combine).
// ---------------------------------------------------------------------------

typedef _Float16 v16h __attribute__((ext_vector_type(16)));
typedef _Float16 v8h  __attribute__((ext_vector_type(8)));
typedef _Float16 v4h  __attribute__((ext_vector_type(4)));
typedef float    v8f  __attribute__((ext_vector_type(8)));
typedef float    v4f  __attribute__((ext_vector_type(4)));

#define DIM       1024     // d_in == d_out
#define NEXP      8
#define NTOK      8192
#define MTILE     32       // tokens per block
#define KSTEPS    32       // 1024 / 32 (global k-steps)
#define NSUB      64       // 1024 / 16 column subtiles
#define KHALF     512      // K staged per phase
#define LN_EPS    1e-5f
#define XPAD      8        // LDS row padding (halfs) to dodge bank conflicts

// ---------------------------------------------------------------------------
// Kernel 1: swizzle W[e][k][n] (f32, row-major over n) into f16 B-fragments.
// B-fragment layout chosen here (contiguous K, A matches; not AMD's table order) (32x16, K x N), wave32:
//   lane L: N = nsub*16 + (L & 15)
//           K = kstep*32 + (L >> 4)*16 + h   for h = 0..15 (contiguous run)
// Storage: wf[(((e*32 + ks)*64 + nsub)*32 + lane)*16 + h]
// ---------------------------------------------------------------------------
__global__ __launch_bounds__(256) void swizzle_w_kernel(
    const float* __restrict__ W, _Float16* __restrict__ wf) {
  const int slot = blockIdx.x * 256 + threadIdx.x;      // 8*32*64*32 = 524288
  if (slot >= NEXP * KSTEPS * NSUB * 32) return;
  const int lane = slot & 31;
  const int nsub = (slot >> 5) & 63;
  const int ks   = (slot >> 11) & 31;
  const int e    = slot >> 16;

  const int n  = nsub * 16 + (lane & 15);
  const int k0 = ks * 32 + (lane >> 4) * 16;
  const float* __restrict__ src = W + ((size_t)e * DIM + k0) * DIM + n;

  v16h v;
#pragma unroll
  for (int h = 0; h < 16; ++h) v[h] = (_Float16)src[(size_t)h * DIM];
  *(v16h*)(wf + (size_t)slot * 16) = v;
}

// ---------------------------------------------------------------------------
// Kernel 2: fused grouped GEMM + bias + ReLU + LayerNorm.
// Grid: 256 blocks (32 tokens each), 256 threads = 8 waves.
// Wave w owns output columns [w*128, w*128+128); rows split into two 16-row
// groups g=0 (rows 0-15) and g=1 (rows 16-31) sharing each B-fragment.
// ---------------------------------------------------------------------------
__global__ __launch_bounds__(256) void moe_fused_kernel(
    const float* __restrict__ x,        // [8192,1024] f32
    const _Float16* __restrict__ wf,    // swizzled f16 weights
    const float* __restrict__ bias,     // [8,1024]
    const float* __restrict__ gamma,    // [8,1024]
    const float* __restrict__ beta,     // [8,1024]
    float* __restrict__ out) {          // [8192,1024] f32

  __shared__ _Float16 sX[MTILE][KHALF + XPAD];  // 32 x 520 f16 (~33 KB)
  __shared__ float sS1[MTILE][8];               // per-(row,wave) partial sums
  __shared__ float sS2[MTILE][8];               // per-(row,wave) partial sumsq
  __shared__ float sMu[MTILE];
  __shared__ float sRs[MTILE];

  const int tid    = threadIdx.x;
  const int lane   = tid & 31;
  const int w      = tid >> 5;        // wave 0..7
  const int m      = lane & 15;       // A row within 16-group / B column
  const int kg     = lane >> 4;       // K-half selector within fragment
  const int blk    = blockIdx.x;
  const int token0 = blk * MTILE;
  const int e      = blk >> 5;        // 32 blocks per expert (1024/32)

  v8f acc[2][8] = {};
  const _Float16* __restrict__ wlane =
      wf + (((size_t)e * KSTEPS) * NSUB + (size_t)w * 8) * 512 + (size_t)lane * 16;

  for (int kh = 0; kh < 2; ++kh) {
    if (kh) __syncthreads();   // all waves done reading previous K-half

    // ---- Stage X[token0..+32, kh*512..+512) into LDS as f16, coalesced ----
#pragma unroll
    for (int i = 0; i < 16; ++i) {
      const int flat = i * 256 + tid;           // float4 id, 0..4095
      const int row  = flat >> 7;               // 0..31
      const int c4   = (flat & 127) * 4;        // 0..508
      v4f xv = *(const v4f*)(x + (size_t)(token0 + row) * DIM + kh * KHALF + c4);
      v4h hv;
      hv[0] = (_Float16)xv[0]; hv[1] = (_Float16)xv[1];
      hv[2] = (_Float16)xv[2]; hv[3] = (_Float16)xv[3];
      *(v4h*)(&sX[row][c4]) = hv;
    }
    __syncthreads();

    // ---- 16 k-steps of this half: 2 A-frags, 8 B-frags, 16 WMMAs each ----
    for (int ks = 0; ks < 16; ++ks) {
      const int kb = ks * 32 + kg * 16;
      // A fragment (contiguous K order, matching swizzled B):
      //   lane kg=0: K {0..15}; kg=1: K {16..31}
      v8h a0lo = *(const v8h*)(&sX[m][kb]);
      v8h a0hi = *(const v8h*)(&sX[m][kb + 8]);
      v8h a1lo = *(const v8h*)(&sX[16 + m][kb]);
      v8h a1hi = *(const v8h*)(&sX[16 + m][kb + 8]);
      v16h a0 = __builtin_shufflevector(a0lo, a0hi,
                                        0, 1, 2, 3, 4, 5, 6, 7,
                                        8, 9, 10, 11, 12, 13, 14, 15);
      v16h a1 = __builtin_shufflevector(a1lo, a1hi,
                                        0, 1, 2, 3, 4, 5, 6, 7,
                                        8, 9, 10, 11, 12, 13, 14, 15);
      const _Float16* __restrict__ wk =
          wlane + (size_t)(kh * 16 + ks) * (NSUB * 512);
#pragma unroll
      for (int s = 0; s < 8; ++s) {
        v16h bf = *(const v16h*)(wk + (size_t)s * 512);
        acc[0][s] = __builtin_amdgcn_wmma_f32_16x16x32_f16(
            false, a0, false, bf, (short)0, acc[0][s], false, false);
        acc[1][s] = __builtin_amdgcn_wmma_f32_16x16x32_f16(
            false, a1, false, bf, (short)0, acc[1][s], false, false);
      }
    }
  }

  // ---- Epilogue: bias + ReLU in registers ----
  // C/D layout: lane(0..15)=N, reg r holds M=r (lanes 0-15) / M=r+8 (16-31).
  float bv[8], gv[8], tv[8];
#pragma unroll
  for (int s = 0; s < 8; ++s) {
    const int ncol = w * 128 + s * 16 + (lane & 15);
    bv[s] = bias[e * DIM + ncol];
    gv[s] = gamma[e * DIM + ncol];
    tv[s] = beta[e * DIM + ncol];
  }
#pragma unroll
  for (int g = 0; g < 2; ++g)
#pragma unroll
    for (int s = 0; s < 8; ++s)
#pragma unroll
      for (int r = 0; r < 8; ++r)
        acc[g][s][r] = fmaxf(acc[g][s][r] + bv[s], 0.0f);

  // ---- Per-row sums over this wave's 128 columns ----
  float s1[2][8], s2[2][8];
#pragma unroll
  for (int g = 0; g < 2; ++g)
#pragma unroll
    for (int r = 0; r < 8; ++r) { s1[g][r] = 0.0f; s2[g][r] = 0.0f; }
#pragma unroll
  for (int g = 0; g < 2; ++g)
#pragma unroll
    for (int s = 0; s < 8; ++s)
#pragma unroll
      for (int r = 0; r < 8; ++r) {
        const float v = acc[g][s][r];
        s1[g][r] += v;
        s2[g][r] += v * v;
      }
  // reduce across the 16-lane half (masks < 16 stay within the half)
#pragma unroll
  for (int mask = 1; mask < 16; mask <<= 1)
#pragma unroll
    for (int g = 0; g < 2; ++g)
#pragma unroll
      for (int r = 0; r < 8; ++r) {
        s1[g][r] += __shfl_xor(s1[g][r], mask, 32);
        s2[g][r] += __shfl_xor(s2[g][r], mask, 32);
      }
  if ((lane & 15) == 0) {
#pragma unroll
    for (int g = 0; g < 2; ++g)
#pragma unroll
      for (int r = 0; r < 8; ++r) {
        const int row = g * 16 + r + 8 * kg;
        sS1[row][w] = s1[g][r];
        sS2[row][w] = s2[g][r];
      }
  }
  __syncthreads();

  // ---- Combine wave partials -> mean / rsqrt(var) per token ----
  if (tid < MTILE) {
    float S1 = 0.0f, S2 = 0.0f;
#pragma unroll
    for (int ww = 0; ww < 8; ++ww) { S1 += sS1[tid][ww]; S2 += sS2[tid][ww]; }
    const float mu  = S1 * (1.0f / (float)DIM);
    const float var = S2 * (1.0f / (float)DIM) - mu * mu;
    sMu[tid] = mu;
    sRs[tid] = rsqrtf(var + LN_EPS);
  }
  __syncthreads();

  // ---- Normalize + affine + store (f32, 64B coalesced segments) ----
#pragma unroll
  for (int g = 0; g < 2; ++g)
#pragma unroll
    for (int s = 0; s < 8; ++s) {
      const int ncol = w * 128 + s * 16 + (lane & 15);
#pragma unroll
      for (int r = 0; r < 8; ++r) {
        const int row = g * 16 + r + 8 * kg;
        const float y = (acc[g][s][r] - sMu[row]) * sRs[row] * gv[s] + tv[s];
        out[(size_t)(token0 + row) * DIM + ncol] = y;
      }
    }
}

// ---------------------------------------------------------------------------
// Inputs (setup_inputs order):
//   0: expert_ordered_input f32 [8192,1024]
//   1: expert_frequency     i64 [8]        (equal capacity -> unused)
//   2: W                    f32 [8,1024,1024]
//   3: b                    f32 [8,1024]
//   4: ln_gamma             f32 [8,1024]
//   5: ln_beta              f32 [8,1024]
// Output: f32 [8192,1024]. Workspace: 16 MB f16 swizzled weights.
// ---------------------------------------------------------------------------
extern "C" void kernel_launch(void* const* d_in, const int* in_sizes, int n_in,
                              void* d_out, int out_size, void* d_ws, size_t ws_size,
                              hipStream_t stream) {
  (void)in_sizes; (void)n_in; (void)out_size; (void)ws_size;
  const float* x     = (const float*)d_in[0];
  const float* W     = (const float*)d_in[2];
  const float* bias  = (const float*)d_in[3];
  const float* gamma = (const float*)d_in[4];
  const float* beta  = (const float*)d_in[5];
  float* out         = (float*)d_out;
  _Float16* wf       = (_Float16*)d_ws;   // 8*1024*1024 halfs = 16 MB

  swizzle_w_kernel<<<dim3(2048), dim3(256), 0, stream>>>(W, wf);
  moe_fused_kernel<<<dim3(NTOK / MTILE), dim3(256), 0, stream>>>(
      x, wf, bias, gamma, beta, out);
}
